// DistanceHead_23965917511883
// MI455X (gfx1250) — matrix operand order repaired and checked by reading, not compiled, not run on hardware
//
#include <hip/hip_runtime.h>
#include <stdint.h>

#define N_NODES 100000LL
#define TPB   256
#define ITEMS 8
#define TILE  (TPB * ITEMS)

typedef __attribute__((ext_vector_type(16))) _Float16     v16h;
typedef __attribute__((ext_vector_type(8)))  float        v8f;
typedef __attribute__((ext_vector_type(4)))  unsigned int v4u;
typedef __attribute__((ext_vector_type(8)))  int          v8i;
typedef __attribute__((ext_vector_type(4)))  int          v4i;

#if defined(__HIP_DEVICE_COMPILE__) && __has_builtin(__builtin_amdgcn_tensor_load_to_lds)
#define USE_TDM 1
#else
#define USE_TDM 0
#endif

#if USE_TDM
typedef __attribute__((address_space(3))) char lds_char;
#endif

// ---------------------------------------------------------------------------
// Kernel 1: fused MLP head.  One wave32 handles a 16-edge tile:
//   h = gelu(dist@W1 + b1); h = LN(h)*gamma+beta; val = h@W2 + b2
// W1 (32x128 f32, 16KB) is staged into LDS by the Tensor Data Mover (one
// TENSOR_LOAD_TO_LDS per block, D# built per CDNA5 ISA),
// then converted to a transposed f16 image [n][k] so each WMMA B fragment is
// two 16-byte LDS runs per lane (2 x ds_load_b128).  8 x v_wmma_f32_16x16x32_f16
// per wave, f32 accumulation; LN stats via half-wave shfl_xor butterflies.
// ---------------------------------------------------------------------------
__global__ __launch_bounds__(256) void k_mlp(
    const float* __restrict__ dist, const float* __restrict__ W1,
    const float* __restrict__ b1,   const float* __restrict__ gamma,
    const float* __restrict__ beta, const float* __restrict__ W2,
    const float* __restrict__ b2,   float* __restrict__ vals, int E)
{
    __shared__ float    w1raw[32 * 128];          // TDM landing buffer (16KB)
    __shared__ _Float16 w1t[128 * 32];            // W1^T as f16: [n][k]
    __shared__ float sb1[128], sg[128], sbt[128], sw2[128];
    const int tid = threadIdx.x;

#if USE_TDM
    if (tid < 32) {
        // --- Tensor DMA descriptor (D#), 2D tile == whole 32x128 f32 tensor.
        // Group0: count=1 | lds_addr | global_addr[56:0] | type=2
        unsigned lds_off = (unsigned)(unsigned long long)(lds_char*)&w1raw[0];
        unsigned long long ga = (unsigned long long)(const void*)W1;
        v4u g0 = { 1u,                                   // count=1 (valid)
                   lds_off,                              // lds_addr (bytes)
                   (unsigned)(ga & 0xffffffffu),         // global_addr[31:0]
                   (unsigned)((ga >> 32) & 0x01ffffffu)  // global_addr[56:32]
                     | (2u << 30) };                     // type=2 ("image")
        // Group1: data_size=2 (4B) | dims/strides in element units
        v8i g1 = { (int)(2u << 16),     // [17:16] data_size=4B; wg_mask=0
                   (int)(128u << 16),   // [79:48] tensor_dim0 = 128 (lo16)
                   (int)(32u  << 16),   // [111:80] tensor_dim1 = 32 (lo16)
                   (int)(128u << 16),   // [127:112] tile_dim0 = 128
                   (int)32,             // [143:128] tile_dim1=32, tile_dim2=0
                   (int)128,            // [207:160] tensor_dim0_stride = 128
                   0, 0 };              // tensor_dim1_stride = 0 (2D)
        v4i gz4 = { 0, 0, 0, 0 };       // groups 2/3 unused for 2D tensors
        v8i gz8 = { 0, 0, 0, 0, 0, 0, 0, 0 };
        // clang-23 / therock-10.0 arity: (g0, g1, g2, g3, g4, cpol)
        __builtin_amdgcn_tensor_load_to_lds(g0, g1, gz4, gz4, gz8, 0);
        __builtin_amdgcn_s_wait_tensorcnt(0);   // s_wait_tensorcnt 0
    }
#else
    for (int idx = tid; idx < 32 * 128; idx += 256) w1raw[idx] = W1[idx];
#endif
    if (tid < 128) { sb1[tid] = b1[tid]; sg[tid] = gamma[tid];
                     sbt[tid] = beta[tid]; sw2[tid] = W2[tid]; }
    __syncthreads();                              // w1raw visible block-wide
    for (int idx = tid; idx < 128 * 32; idx += 256) {
        int k = idx >> 7, n = idx & 127;          // w1raw is [32][128] row-major
        w1t[n * 32 + k] = (_Float16)w1raw[idx];
    }
    __syncthreads();

    const float b2s  = b2[0];
    const int wave = tid >> 5, lane = tid & 31;
    const int h = lane >> 4, n = lane & 15;       // half-wave, lane-in-half
    const long long edge0 = (long long)blockIdx.x * 128 + wave * 16;

    long long erow = edge0 + n; if (erow >= E) erow = (long long)E - 1;
    const float* arow = dist + erow * 32;
    __builtin_prefetch(arow, 0, 3);               // global_prefetch_b8

    // A fragment: 16-bit A 16x32 layout. VGPR i holds K pair:
    //   i<4 : k0 = 2i + 8h     i>=4 : k0 = 16 + 2(i-4) + 8h
    v16h a;
#pragma unroll
    for (int i = 0; i < 8; i++) {
        int k0 = (i < 4 ? 2 * i : 16 + 2 * (i - 4)) + 8 * h;
        float2 f = *(const float2*)(arow + k0);
        a[2 * i]     = (_Float16)f.x;
        a[2 * i + 1] = (_Float16)f.y;
    }

    float acc[8][8];                              // [ntile][row-vgpr]
#pragma unroll
    for (int nt = 0; nt < 8; nt++) {
        // B fragment: 16-bit B 32x16 layout. VGPR i holds K pair (i<4 ? 2i : 16+2(i-4)) + 8h.
        const _Float16* bcol = &w1t[(nt * 16 + n) * 32];
        v16h b;
#pragma unroll
        for (int i = 0; i < 8; i++) {
            int k0 = (i < 4 ? 2 * i : 16 + 2 * (i - 4)) + 8 * h;
            b[2 * i]     = bcol[k0];
            b[2 * i + 1] = bcol[k0 + 1];
        }
        v8f c = {};
        c = __builtin_amdgcn_wmma_f32_16x16x32_f16(false, a, false, b,
                                                   (short)0, c, false, false);
#pragma unroll
        for (int j = 0; j < 8; j++) acc[nt][j] = c[j];
    }

    // bias + exact (erf) GELU + LN stats. C/D layout: VGPR j is row j+8h, col n.
    float s[8], q[8];
#pragma unroll
    for (int j = 0; j < 8; j++) { s[j] = 0.f; q[j] = 0.f; }
#pragma unroll
    for (int nt = 0; nt < 8; nt++) {
        int col = nt * 16 + n;
        float bb = sb1[col];
#pragma unroll
        for (int j = 0; j < 8; j++) {
            float e = acc[nt][j] + bb;
            float g = 0.5f * e * (1.0f + erff(e * 0.70710678118654752f));
            acc[nt][j] = g; s[j] += g; q[j] += g * g;
        }
    }
#pragma unroll
    for (int j = 0; j < 8; j++) {
#pragma unroll
        for (int msk = 1; msk < 16; msk <<= 1) {  // butterfly within 16 lanes
            s[j] += __shfl_xor(s[j], msk, 32);
            q[j] += __shfl_xor(q[j], msk, 32);
        }
    }
#pragma unroll
    for (int j = 0; j < 8; j++) {
        float mu  = s[j] * (1.0f / 128.0f);
        float var = q[j] * (1.0f / 128.0f) - mu * mu;
        float rs  = rsqrtf(var + 1e-5f);
        float t = 0.f;
#pragma unroll
        for (int nt = 0; nt < 8; nt++) {
            int col = nt * 16 + n;
            float y = (acc[nt][j] - mu) * rs * sg[col] + sbt[col];
            t += y * sw2[col];
        }
#pragma unroll
        for (int msk = 1; msk < 16; msk <<= 1) t += __shfl_xor(t, msk, 32);
        long long eo = edge0 + j + 8 * h;
        if (n == 0 && eo < E) vals[eo] = t + b2s;
    }
}

// ---------------------------------------------------------------------------
// Kernel 2: build (key,value) pairs for both edge directions.
// ---------------------------------------------------------------------------
__global__ void k_pairs(const long long* __restrict__ ei,
                        const float* __restrict__ vals,
                        unsigned long long* __restrict__ keys,
                        float* __restrict__ pv, int E)
{
    int i = blockIdx.x * blockDim.x + threadIdx.x;
    if (i >= E) return;
    long long r = ei[i], c = ei[E + i];
    float v = vals[i];
    keys[i]     = (unsigned long long)(r * N_NODES + c); pv[i]     = v;
    keys[E + i] = (unsigned long long)(c * N_NODES + r); pv[E + i] = v;
}

// ---------------------------------------------------------------------------
// LSD radix sort, 8-bit digits x 5 passes (keys < 2^34).
// ---------------------------------------------------------------------------
__global__ __launch_bounds__(256) void k_hist(
    const unsigned long long* __restrict__ keys, int n, int shift,
    unsigned* __restrict__ hist, int NB)
{
    __shared__ unsigned lh[256];
    int tid = threadIdx.x;
    lh[tid] = 0; __syncthreads();
    long long start = (long long)blockIdx.x * TILE;
    for (int j = 0; j < ITEMS; j++) {
        long long e = start + (long long)j * TPB + tid;
        if (e < n) {
            unsigned d = (unsigned)((keys[e] >> shift) & 255);
            atomicAdd(&lh[d], 1u);
        }
    }
    __syncthreads();
    hist[(size_t)tid * NB + blockIdx.x] = lh[tid];
}

// Per-digit exclusive scan over blocks (grid = 256, block d = digit).
// Leaves within-digit exclusive prefixes in `hist`, digit totals in `dtot`.
__global__ __launch_bounds__(256) void k_scan_digit(
    unsigned* __restrict__ hist, unsigned* __restrict__ dtot, int NB)
{
    __shared__ unsigned sh[256];
    int d = blockIdx.x, tid = threadIdx.x;
    unsigned carry = 0;
    for (int base = 0; base < NB; base += 256) {
        int idx = base + tid;
        unsigned v = (idx < NB) ? hist[(size_t)d * NB + idx] : 0u;
        sh[tid] = v; __syncthreads();
        for (int off = 1; off < 256; off <<= 1) {
            unsigned t = (tid >= off) ? sh[tid - off] : 0u;
            __syncthreads(); sh[tid] += t; __syncthreads();
        }
        unsigned incl = sh[tid], tot = sh[255];
        if (idx < NB) hist[(size_t)d * NB + idx] = carry + incl - v;
        carry += tot;
        __syncthreads();
    }
    if (tid == 0) dtot[d] = carry;
}

// Generic single-block in-place exclusive scan of `arr[0..len)`, total -> *tot.
__global__ __launch_bounds__(256) void k_scan_one(
    unsigned* __restrict__ arr, int len, unsigned* __restrict__ tot)
{
    __shared__ unsigned sh[256];
    int tid = threadIdx.x;
    unsigned carry = 0;
    for (int base = 0; base < len; base += 256) {
        int idx = base + tid;
        unsigned v = (idx < len) ? arr[idx] : 0u;
        sh[tid] = v; __syncthreads();
        for (int off = 1; off < 256; off <<= 1) {
            unsigned t = (tid >= off) ? sh[tid - off] : 0u;
            __syncthreads(); sh[tid] += t; __syncthreads();
        }
        unsigned incl = sh[tid], ctot = sh[255];
        if (idx < len) arr[idx] = carry + incl - v;
        carry += ctot;
        __syncthreads();
    }
    if (tid == 0) tot[0] = carry;
}

// Stable scatter: wave32 ballot digit-match masks give in-wave stable ranks;
// LDS per-wave digit counts + per-digit running base give block-stable ranks.
__global__ __launch_bounds__(256) void k_scatter(
    const unsigned long long* __restrict__ keys, const float* __restrict__ pv,
    unsigned long long* __restrict__ okeys, float* __restrict__ opv,
    int n, int shift, const unsigned* __restrict__ hist,
    const unsigned* __restrict__ dstart, int NB)
{
    __shared__ unsigned running[256];
    __shared__ unsigned wh[8 * 256];
    int tid = threadIdx.x;
    int lane = tid & 31, wv = tid >> 5;
    unsigned lt = (1u << lane) - 1u;
    running[tid] = 0;
    __syncthreads();
    long long start = (long long)blockIdx.x * TILE;
    for (int j = 0; j < ITEMS; j++) {
#pragma unroll
        for (int w = 0; w < 8; w++) wh[w * 256 + tid] = 0;
        __syncthreads();
        long long e = start + (long long)j * TPB + tid;
        bool valid = e < n;
        unsigned long long key = 0; float v = 0.f; unsigned d = 0;
        if (valid) { key = keys[e]; v = pv[e];
                     d = (unsigned)((key >> shift) & 255); }
        unsigned m = (unsigned)__ballot(valid);
#pragma unroll
        for (int bit = 0; bit < 8; bit++) {
            unsigned bb = (unsigned)__ballot(valid && ((d >> bit) & 1u));
            m &= ((d >> bit) & 1u) ? bb : ~bb;
        }
        unsigned wrank = __popc(m & lt);
        bool leader = valid && ((m & lt) == 0u);
        if (leader) wh[wv * 256 + d] = (unsigned)__popc(m);
        __syncthreads();
        // thread tid owns digit tid: prefix across the 8 waves + running base
        unsigned p = running[tid];
#pragma unroll
        for (int w = 0; w < 8; w++) {
            unsigned t = wh[w * 256 + tid];
            wh[w * 256 + tid] = p; p += t;
        }
        running[tid] = p;
        __syncthreads();
        if (valid) {
            unsigned localRank = wh[wv * 256 + d] + wrank;
            unsigned pos = dstart[d] + hist[(size_t)d * NB + blockIdx.x] + localRank;
            okeys[pos] = key; opv[pos] = v;
        }
        __syncthreads();
    }
}

// ---------------------------------------------------------------------------
// Segment (unique) machinery over the sorted array.
// ---------------------------------------------------------------------------
__global__ __launch_bounds__(256) void k_flag_count(
    const unsigned long long* __restrict__ keys, int n, unsigned* __restrict__ bc)
{
    __shared__ unsigned sh[256];
    int tid = threadIdx.x;
    long long start = (long long)blockIdx.x * TILE;
    unsigned c = 0;
    for (int j = 0; j < ITEMS; j++) {
        long long e = start + (long long)j * TPB + tid;
        if (e < n) c += (e == 0) || (keys[e] != keys[e - 1]);
    }
    sh[tid] = c; __syncthreads();
    for (int off = 128; off > 0; off >>= 1) {
        if (tid < off) sh[tid] += sh[tid + off];
        __syncthreads();
    }
    if (tid == 0) bc[blockIdx.x] = sh[0];
}

__global__ void k_zero(float* __restrict__ sums, unsigned* __restrict__ cnts, int n)
{
    long long i = (long long)blockIdx.x * blockDim.x + threadIdx.x;
    if (i < n) { sums[i] = 0.f; cnts[i] = 0u; }
}

__global__ __launch_bounds__(256) void k_segreduce(
    const unsigned long long* __restrict__ keys, const float* __restrict__ pv,
    int n, const unsigned* __restrict__ blockOff,
    unsigned long long* __restrict__ uniqKeys,
    float* __restrict__ sums, unsigned* __restrict__ cnts)
{
    __shared__ unsigned sh[256];
    int tid = threadIdx.x;
    long long start = (long long)blockIdx.x * TILE;
    unsigned carry = blockOff[blockIdx.x];
    for (int j = 0; j < ITEMS; j++) {
        long long e = start + (long long)j * TPB + tid;
        bool valid = e < n;
        unsigned long long key = 0; float v = 0.f; unsigned flag = 0;
        if (valid) { key = keys[e]; v = pv[e];
                     flag = (e == 0) || (key != keys[e - 1]); }
        sh[tid] = flag; __syncthreads();
        for (int off = 1; off < 256; off <<= 1) {
            unsigned t = (tid >= off) ? sh[tid - off] : 0u;
            __syncthreads(); sh[tid] += t; __syncthreads();
        }
        unsigned incl = sh[tid], tot = sh[255];
        if (valid) {
            unsigned seg = carry + incl - 1;     // global segment id of e
            if (flag) uniqKeys[seg] = key;
            atomicAdd(&sums[seg], v);
            atomicAdd(&cnts[seg], 1u);
        }
        carry += tot;
        __syncthreads();
    }
}

// Final output. Flat float layout: [dist_out (n) | ei_row (n) | ei_col (n)].
// Node ids < 2^24 so f32 holds them exactly; padding mirrors numpy semantics
// for fill=-1:  -1//N_NODES == -1,  -1 % N_NODES == 99999.
__global__ void k_out(const unsigned long long* __restrict__ uniqKeys,
                      const float* __restrict__ sums,
                      const unsigned* __restrict__ cnts,
                      const unsigned* __restrict__ total,
                      float* __restrict__ out, int n)
{
    long long u = (long long)blockIdx.x * blockDim.x + threadIdx.x;
    if (u >= n) return;
    unsigned U = total[0];
    float dv; long long r, c;
    if (u < (long long)U) {
        long long key = (long long)uniqKeys[u];
        dv = sums[u] / fmaxf((float)cnts[u], 1.0f);
        r = key / N_NODES; c = key % N_NODES;
    } else {
        dv = 0.f; r = -1; c = 99999;
    }
    out[u]           = dv;
    out[n + u]       = (float)r;
    out[2LL * n + u] = (float)c;
}

// ---------------------------------------------------------------------------
extern "C" void kernel_launch(void* const* d_in, const int* in_sizes, int n_in,
                              void* d_out, int out_size, void* d_ws, size_t ws_size,
                              hipStream_t stream)
{
    (void)n_in; (void)out_size; (void)ws_size;
    const float*     dist = (const float*)d_in[0];
    const long long* ei   = (const long long*)d_in[1];
    const float*     W1   = (const float*)d_in[2];
    const float*     b1   = (const float*)d_in[3];
    const float*     gm   = (const float*)d_in[4];
    const float*     bt   = (const float*)d_in[5];
    const float*     W2   = (const float*)d_in[6];
    const float*     b2   = (const float*)d_in[7];

    const int E  = in_sizes[0] / 32;
    const int n  = 2 * E;
    const int NB = (n + TILE - 1) / TILE;

    // workspace carve-out (~170 MB for E=2M)
    char* p = (char*)d_ws;
    auto carve = [&](size_t bytes) -> char* {
        char* r = p; p += (bytes + 255) & ~(size_t)255; return r;
    };
    float*              vals  = (float*)carve((size_t)E * 4);
    unsigned long long* keysA = (unsigned long long*)carve((size_t)n * 8);
    unsigned long long* keysB = (unsigned long long*)carve((size_t)n * 8);
    float*              pvA   = (float*)carve((size_t)n * 4);
    float*              pvB   = (float*)carve((size_t)n * 4);
    unsigned*           hist  = (unsigned*)carve((size_t)256 * NB * 4);
    unsigned*           dst   = (unsigned*)carve(256 * 4);   // digit totals->starts
    unsigned*           bcnt  = (unsigned*)carve((size_t)NB * 4);
    unsigned*           totU  = (unsigned*)carve(256);
    unsigned long long* uniqK = (unsigned long long*)carve((size_t)n * 8);
    float*              sums  = (float*)carve((size_t)n * 4);
    unsigned*           cnts  = (unsigned*)carve((size_t)n * 4);

    // 1) fused MLP head (TDM staging + WMMA)
    k_mlp<<<(E + 127) / 128, 256, 0, stream>>>(dist, W1, b1, gm, bt, W2, b2, vals, E);

    // 2) build undirected pairs
    k_pairs<<<(E + 255) / 256, 256, 0, stream>>>(ei, vals, keysA, pvA, E);

    // 3) LSD radix sort: 5 x 8-bit passes (keys < 2^34)
    unsigned long long *ks = keysA, *kd = keysB;
    float *vs = pvA, *vd = pvB;
    for (int pass = 0; pass < 5; pass++) {
        int shift = 8 * pass;
        k_hist      <<<NB,  256, 0, stream>>>(ks, n, shift, hist, NB);
        k_scan_digit<<<256, 256, 0, stream>>>(hist, dst, NB);
        k_scan_one  <<<1,   256, 0, stream>>>(dst, 256, totU);   // digit starts
        k_scatter   <<<NB,  256, 0, stream>>>(ks, vs, kd, vd, n, shift, hist, dst, NB);
        unsigned long long* tk = ks; ks = kd; kd = tk;
        float* tv = vs; vs = vd; vd = tv;
    }
    // sorted data now in (ks, vs) == (keysB, pvB)

    // 4) unique + segment mean
    k_zero      <<<(n + 255) / 256, 256, 0, stream>>>(sums, cnts, n);
    k_flag_count<<<NB, 256, 0, stream>>>(ks, n, bcnt);
    k_scan_one  <<<1,  256, 0, stream>>>(bcnt, NB, totU);        // totU[0] = U
    k_segreduce <<<NB, 256, 0, stream>>>(ks, vs, n, bcnt, uniqK, sums, cnts);

    // 5) outputs
    k_out<<<(n + 255) / 256, 256, 0, stream>>>(uniqK, sums, cnts, totU,
                                               (float*)d_out, n);
}
